// EnhancedTransformerBlock_78451872629408
// MI455X (gfx1250) — matrix operand order repaired and checked by reading, not compiled, not run on hardware
//
#include <hip/hip_runtime.h>

// MI455X / gfx1250 transformer block: bf16 WMMA + async global->LDS staging.
// Pipeline: pack weights -> LN1 -> qkv GEMM (bf16 out) -> WMMA flash attention
//        -> proj GEMM (+residual) -> LN2 -> gate GEMM -> up GEMM (fused SiLU*up, bf16)
//        -> down GEMM (+residual) -> d_out.

typedef __bf16 bf16_t;
typedef __attribute__((ext_vector_type(16))) __bf16 v16bf;
typedef __attribute__((ext_vector_type(8)))  __bf16 v8bf;
typedef __attribute__((ext_vector_type(8)))  float  v8f;

#define WMMA_BF16(a, b, c) \
  __builtin_amdgcn_wmma_f32_16x16x32_bf16(false, (a), false, (b), (short)0, (c), false, false)

// Async global -> LDS copy of 16 bytes per lane (CDNA5 path, ASYNCcnt-tracked).
// INST_OFFSET applies to both the LDS and global addresses (ISA 15.18), so one
// base pair serves several copies via immediate offsets.
#define ASYNC_B128(ldsp, gp, OFF)                                              \
  asm volatile("global_load_async_to_lds_b128 %0, %1, off offset:%2"           \
               :: "v"((unsigned)(uintptr_t)(ldsp)), "v"(gp), "i"(OFF) : "memory")

__device__ __forceinline__ void wait_async_5() {
  asm volatile("s_wait_asynccnt 0x5" ::: "memory");
}
__device__ __forceinline__ void wait_async_0() {
  asm volatile("s_wait_asynccnt 0x0" ::: "memory");
}

__device__ __forceinline__ v16bf join16(v8bf lo, v8bf hi) {
  return __builtin_shufflevector(lo, hi, 0,1,2,3,4,5,6,7,8,9,10,11,12,13,14,15);
}

// A-fragment (16xK tile, 16-bit): contiguous k order (matches B fragments).
// lanes 0-15 : M=lane,    elems 0..15 = K k0..k0+15
// lanes 16-31: M=lane-16, elems 0..15 = K k0+16..k0+31
__device__ __forceinline__ v16bf load_afrag(const bf16_t* A, int ld, int row0, int k0, int lane) {
  int m = row0 + (lane & 15);
  int koff = k0 + ((lane & 16) ? 16 : 0);
  const bf16_t* p = A + (size_t)m * ld + koff;
  v8bf lo = *(const v8bf*)(p);
  v8bf hi = *(const v8bf*)(p + 8);
  return join16(lo, hi);
}

// B-fragment (32x16 tile, 16-bit) from a column-contiguous source:
// element (k,n) lives at S[n*ld + k].
__device__ __forceinline__ v16bf load_bfrag_cm(const bf16_t* S, int ld, int k0, int n0, int lane) {
  int n = n0 + (lane & 15);
  int k = k0 + ((lane & 16) ? 16 : 0);
  const bf16_t* p = S + (size_t)n * ld + k;
  v8bf lo = *(const v8bf*)(p);
  v8bf hi = *(const v8bf*)(p + 8);
  return join16(lo, hi);
}

// Packed B fragment (32 lanes x 16 bf16 contiguous) at fragment slot `nt`.
__device__ __forceinline__ v16bf load_bpack_at(const bf16_t* base, int nt, int lane) {
  const bf16_t* p = base + (size_t)nt * 512 + (size_t)lane * 16;
  v8bf lo = *(const v8bf*)(p);
  v8bf hi = *(const v8bf*)(p + 8);
  return join16(lo, hi);
}

// ---------------- weight packing: fp32 [K x N] -> bf16 B-fragment-major ----
__global__ void pack_b_kernel(const float* __restrict__ W, bf16_t* __restrict__ out,
                              int K, int N, int KT, int NT) {
  int t = blockIdx.x * blockDim.x + threadIdx.x;
  int total = KT * NT * 32;
  if (t >= total) return;
  int lane = t & 31;
  int frag = t >> 5;
  int nt = frag % NT;
  int kt = frag / NT;
  int n = nt * 16 + (lane & 15);
  int kb = kt * 32 + ((lane & 16) ? 16 : 0);
  bf16_t* dst = out + (size_t)frag * 512 + (size_t)lane * 16;
#pragma unroll
  for (int j = 0; j < 16; ++j) {
    int k = kb + j;
    float v = (k < K && n < N) ? W[(size_t)k * N + n] : 0.f;
    dst[j] = (bf16_t)v;
  }
}

// ---------------- layernorm (fp32 in) -> bf16 row ---------------------------
__global__ __launch_bounds__(256)
void ln_to_bf16_kernel(const float* __restrict__ x, const float* __restrict__ g,
                       const float* __restrict__ bta, bf16_t* __restrict__ out, int C) {
  int row = blockIdx.x;
  int tid = threadIdx.x;
  const float* xr = x + (size_t)row * C;
  __shared__ float red[256];
  float s = 0.f;
  for (int i = tid; i < C; i += 256) s += xr[i];
  red[tid] = s;
  __syncthreads();
  for (int st = 128; st > 0; st >>= 1) {
    if (tid < st) red[tid] += red[tid + st];
    __syncthreads();
  }
  float mu = red[0] / (float)C;
  __syncthreads();
  float vs = 0.f;
  for (int i = tid; i < C; i += 256) { float d = xr[i] - mu; vs += d * d; }
  red[tid] = vs;
  __syncthreads();
  for (int st = 128; st > 0; st >>= 1) {
    if (tid < st) red[tid] += red[tid + st];
    __syncthreads();
  }
  float rstd = rsqrtf(red[0] / (float)C + 1e-5f);
  bf16_t* orow = out + (size_t)row * C;
  for (int i = tid; i < C; i += 256)
    orow[i] = (bf16_t)((xr[i] - mu) * rstd * g[i] + bta[i]);
}

// ---------------- WMMA GEMM with async double-buffered LDS staging ----------
// C[M x Np] = A[M x Kp](bf16) @ Bpack + bias, epilogue by mode:
//  0: outF = acc+bias            1: outH = bf16(acc+bias)
//  2: outF = acc+bias+extraF     3: outH = bf16(silu(extraF)*(acc+bias))
// Block tile 256(M) x 64(N); 8 waves as 4(M) x 2(N); wave tile 64x32 = 8 WMMA
// per k-step. A tile 256x32 (16KB) + 4 packed B frags (4KB) staged per k-step
// via global_load_async_to_lds_b128, double buffered, overlapped with WMMA.
__global__ __launch_bounds__(256)
void gemm_bf16_wmma(const bf16_t* __restrict__ A, const bf16_t* __restrict__ Bp,
                    const float* __restrict__ bias, int Kp, int Np, int Nbias,
                    float* __restrict__ outF, bf16_t* __restrict__ outH,
                    const float* __restrict__ extraF, int mode) {
  __shared__ alignas(16) bf16_t Alds[2][256 * 32];
  __shared__ alignas(16) bf16_t Blds[2][4 * 512];

  int tid = threadIdx.x;
  int lane = tid & 31;
  int w = tid >> 5;
  int wm = w & 3, wn = w >> 2;
  int mblk = blockIdx.y * 256;
  int nbase = blockIdx.x * 64 + wn * 32;
  int NT = Np >> 4;
  int KT = Kp >> 5;
  int ntBlk = blockIdx.x * 4;

  // staging: thread owns one 32-elem A row (64B = 4 async B128s) + 16B of B
  const bf16_t* gArow = A + (size_t)(mblk + tid) * Kp;
  bf16_t* lArow0 = &Alds[0][tid * 32];
  bf16_t* lArow1 = &Alds[1][tid * 32];
  bf16_t* lBrow0 = &Blds[0][tid * 8];
  bf16_t* lBrow1 = &Blds[1][tid * 8];

  auto stage = [&](int kt, int buf) {
    const bf16_t* gA = gArow + (kt << 5);
    bf16_t* lA = buf ? lArow1 : lArow0;
    ASYNC_B128(lA, gA, 0);
    ASYNC_B128(lA, gA, 16);
    ASYNC_B128(lA, gA, 32);
    ASYNC_B128(lA, gA, 48);
    const bf16_t* gB = Bp + ((size_t)kt * NT + ntBlk) * 512 + (size_t)tid * 8;
    ASYNC_B128(buf ? lBrow1 : lBrow0, gB, 0);
  };

  v8f acc[4][2];
#pragma unroll
  for (int i = 0; i < 4; ++i)
#pragma unroll
    for (int j = 0; j < 2; ++j)
#pragma unroll
      for (int e = 0; e < 8; ++e) acc[i][j][e] = 0.f;

  auto compute = [&](int buf) {
    const bf16_t* Ab = Alds[buf];
    const bf16_t* Bb = Blds[buf];
    v16bf b0 = load_bpack_at(Bb, wn * 2, lane);
    v16bf b1 = load_bpack_at(Bb, wn * 2 + 1, lane);
#pragma unroll
    for (int mi = 0; mi < 4; ++mi) {
      v16bf a = load_afrag(Ab, 32, wm * 64 + mi * 16, 0, lane);
      acc[mi][0] = WMMA_BF16(a, b0, acc[mi][0]);
      acc[mi][1] = WMMA_BF16(a, b1, acc[mi][1]);
    }
  };

  // branch-free steady state, peeled tail
  stage(0, 0);
  for (int kt = 0; kt < KT - 1; ++kt) {
    int buf = kt & 1;
    stage(kt + 1, buf ^ 1);
    wait_async_5();   // previous stage landed; next stage (5 ops) in flight
    __syncthreads();
    compute(buf);
    __syncthreads();  // everyone done reading buf before it is restaged
  }
  wait_async_0();
  __syncthreads();
  compute((KT - 1) & 1);

  int rbump = (lane & 16) ? 8 : 0;
#pragma unroll
  for (int mi = 0; mi < 4; ++mi)
#pragma unroll
    for (int ni = 0; ni < 2; ++ni) {
      int col = nbase + ni * 16 + (lane & 15);
      float bv = (col < Nbias) ? bias[col] : 0.f;
#pragma unroll
      for (int r = 0; r < 8; ++r) {
        int row = mblk + wm * 64 + mi * 16 + r + rbump;
        size_t idx = (size_t)row * Np + col;
        float v = acc[mi][ni][r] + bv;
        if (mode == 0) {
          outF[idx] = v;
        } else if (mode == 1) {
          outH[idx] = (bf16_t)v;
        } else if (mode == 2) {
          outF[idx] = v + extraF[idx];
        } else {
          float gt = extraF[idx];
          float sg = gt / (1.f + __expf(-gt));
          outH[idx] = (bf16_t)(sg * v);
        }
      }
    }
}

// ---------------- WMMA flash attention (ALiBi + causal) --------------------
// One wave per (b, h, 16-query tile). 32-key blocks; QK^T and PV via WMMA.
__global__ __launch_bounds__(32)
void attn_wmma_kernel(const bf16_t* __restrict__ qkv, bf16_t* __restrict__ out,
                      int T, int H) {
  const int C3 = 3072, Cc = 1024, D = 64;
  int lane = threadIdx.x;
  int QT = T >> 4;
  int blk = blockIdx.x;
  int qt = blk % QT;
  int bh = blk / QT;
  int h = bh % H;
  int b = bh / H;
  const bf16_t* Q = qkv + (size_t)b * T * C3 + h * D;
  const bf16_t* K = Q + Cc;
  const bf16_t* V = Q + 2 * Cc;

  __shared__ alignas(16) bf16_t Plds[16 * 32];  // P tile, row-major 16x32
  __shared__ alignas(16) bf16_t Vt[64 * 32];    // V tile transposed: Vt[d*32 + key]

  int q0 = qt * 16;
  int qhi = q0 + 15;
  float slope = exp2f(-0.5f * (float)(h + 1));
  int rbump = (lane & 16) ? 8 : 0;
  int nlane = lane & 15;

  v16bf qf0 = load_afrag(Q, C3, q0, 0, lane);
  v16bf qf1 = load_afrag(Q, C3, q0, 32, lane);

  v8f o[4];
  float m_run[8], l_run[8];
#pragma unroll
  for (int r = 0; r < 8; ++r) { m_run[r] = -1e30f; l_run[r] = 0.f; }
#pragma unroll
  for (int nt = 0; nt < 4; ++nt)
#pragma unroll
    for (int r = 0; r < 8; ++r) o[nt][r] = 0.f;

  for (int kb = 0; kb <= qhi; kb += 32) {
    // prefetch next key/value block (global_prefetch_b8)
    if (kb + 32 <= qhi) {
      __builtin_prefetch(K + (size_t)(kb + 32 + lane) * C3, 0, 1);
      __builtin_prefetch(V + (size_t)(kb + 32 + lane) * C3, 0, 1);
    }

    // Stage V tile transposed into LDS: lane owns key row kb+lane.
    const bf16_t* vrow = V + (size_t)(kb + lane) * C3;
#pragma unroll
    for (int d8 = 0; d8 < 8; ++d8) {
      v8bf vv = *(const v8bf*)(vrow + d8 * 8);
#pragma unroll
      for (int e = 0; e < 8; ++e) Vt[(d8 * 8 + e) * 32 + lane] = vv[e];
    }

    // S = Q K^T for 2 key sub-tiles of 16
    v8f s[2];
#pragma unroll
    for (int t = 0; t < 2; ++t) {
#pragma unroll
      for (int e = 0; e < 8; ++e) s[t][e] = 0.f;
      v16bf bk0 = load_bfrag_cm(K, C3, 0, kb + t * 16, lane);
      v16bf bk1 = load_bfrag_cm(K, C3, 32, kb + t * 16, lane);
      s[t] = WMMA_BF16(qf0, bk0, s[t]);
      s[t] = WMMA_BF16(qf1, bk1, s[t]);
    }

    // scale + ALiBi (+slope*|dq-dk| per reference) + causal mask
#pragma unroll
    for (int t = 0; t < 2; ++t) {
      int key = kb + t * 16 + nlane;
#pragma unroll
      for (int r = 0; r < 8; ++r) {
        int row = q0 + r + rbump;
        float v = s[t][r] * 0.125f + slope * fabsf((float)(row - key));
        s[t][r] = (key <= row) ? v : -1e30f;
      }
    }

    // online softmax: row r lives across the 16 lanes of each half-wave
    float scl[8];
#pragma unroll
    for (int r = 0; r < 8; ++r) {
      float mx = fmaxf(s[0][r], s[1][r]);
#pragma unroll
      for (int off = 1; off < 16; off <<= 1) mx = fmaxf(mx, __shfl_xor(mx, off, 32));
      float mnew = fmaxf(m_run[r], mx);
      scl[r] = __expf(m_run[r] - mnew);
      m_run[r] = mnew;
      float p0 = __expf(s[0][r] - mnew);
      float p1 = __expf(s[1][r] - mnew);
      s[0][r] = p0;
      s[1][r] = p1;
      float sum = p0 + p1;
#pragma unroll
      for (int off = 1; off < 16; off <<= 1) sum += __shfl_xor(sum, off, 32);
      l_run[r] = l_run[r] * scl[r] + sum;
    }
#pragma unroll
    for (int nt = 0; nt < 4; ++nt)
#pragma unroll
      for (int r = 0; r < 8; ++r) o[nt][r] *= scl[r];

    // P (C-layout) -> LDS -> A-fragment layout
#pragma unroll
    for (int t = 0; t < 2; ++t)
#pragma unroll
      for (int r = 0; r < 8; ++r)
        Plds[(r + rbump) * 32 + t * 16 + nlane] = (bf16_t)s[t][r];
    __syncthreads();
    v16bf pf = load_afrag(Plds, 32, 0, 0, lane);
#pragma unroll
    for (int nt = 0; nt < 4; ++nt) {
      v16bf bv = load_bfrag_cm(Vt, 32, 0, nt * 16, lane);
      o[nt] = WMMA_BF16(pf, bv, o[nt]);
    }
    __syncthreads();
  }

  // normalize and store bf16 into attn-out [B*T x C] at head slice
#pragma unroll
  for (int nt = 0; nt < 4; ++nt)
#pragma unroll
    for (int r = 0; r < 8; ++r) {
      int row = q0 + r + rbump;
      out[(size_t)(b * T + row) * Cc + h * D + nt * 16 + nlane] =
          (bf16_t)(o[nt][r] / l_run[r]);
    }
}

// ---------------- host orchestration ---------------------------------------
extern "C" void kernel_launch(void* const* d_in, const int* in_sizes, int n_in,
                              void* d_out, int out_size, void* d_ws, size_t ws_size,
                              hipStream_t stream) {
  (void)in_sizes; (void)n_in; (void)out_size; (void)ws_size;
  const int Bb = 2, T = 2048, C = 1024, H = 16;
  const int M = Bb * T;       // 4096
  const int NQKV = 3 * C;     // 3072
  const int HID = 2730, HIDP = 2752;  // padded to /32

  const float* x      = (const float*)d_in[0];
  const float* ln1_g  = (const float*)d_in[1];
  const float* ln1_b  = (const float*)d_in[2];
  const float* qkv_w  = (const float*)d_in[3];
  const float* qkv_b  = (const float*)d_in[4];
  const float* proj_w = (const float*)d_in[5];
  const float* proj_b = (const float*)d_in[6];
  const float* ln2_g  = (const float*)d_in[7];
  const float* ln2_b  = (const float*)d_in[8];
  const float* gate_w = (const float*)d_in[9];
  const float* gate_b = (const float*)d_in[10];
  const float* up_w   = (const float*)d_in[11];
  const float* up_b   = (const float*)d_in[12];
  const float* down_w = (const float*)d_in[13];
  const float* down_b = (const float*)d_in[14];
  float* out = (float*)d_out;

  char* wp = (char*)d_ws;
  auto take = [&wp](size_t bytes) {
    char* p = wp;
    wp += (bytes + 255) & ~(size_t)255;
    return p;
  };
  bf16_t* bpQKV  = (bf16_t*)take((size_t)C * NQKV * 2);
  bf16_t* bpProj = (bf16_t*)take((size_t)C * C * 2);
  bf16_t* bpGate = (bf16_t*)take((size_t)C * HIDP * 2);
  bf16_t* bpUp   = (bf16_t*)take((size_t)C * HIDP * 2);
  bf16_t* bpDown = (bf16_t*)take((size_t)HIDP * C * 2);
  bf16_t* abf    = (bf16_t*)take((size_t)M * C * 2);
  bf16_t* qkvb   = (bf16_t*)take((size_t)M * NQKV * 2);
  bf16_t* attnb  = (bf16_t*)take((size_t)M * C * 2);
  float*  x2     = (float*)take((size_t)M * C * 4);
  float*  gateF  = (float*)take((size_t)M * HIDP * 4);
  bf16_t* hffb   = (bf16_t*)take((size_t)M * HIDP * 2);

  auto pack = [&](const float* W, bf16_t* dst, int K, int N, int KT, int NT) {
    int total = KT * NT * 32;
    pack_b_kernel<<<(total + 255) / 256, 256, 0, stream>>>(W, dst, K, N, KT, NT);
  };
  pack(qkv_w,  bpQKV,  C,   NQKV, C / 32,    NQKV / 16);
  pack(proj_w, bpProj, C,   C,    C / 32,    C / 16);
  pack(gate_w, bpGate, C,   HID,  C / 32,    HIDP / 16);
  pack(up_w,   bpUp,   C,   HID,  C / 32,    HIDP / 16);
  pack(down_w, bpDown, HID, C,    HIDP / 32, C / 16);

  // LN1 -> bf16
  ln_to_bf16_kernel<<<M, 256, 0, stream>>>(x, ln1_g, ln1_b, abf, C);
  // qkv = LN1 @ W_qkv + b  (bf16 out)
  gemm_bf16_wmma<<<dim3(NQKV / 64, M / 256), 256, 0, stream>>>(
      abf, bpQKV, qkv_b, C, NQKV, NQKV, nullptr, qkvb, nullptr, 1);
  // attention (ALiBi + causal, flash-style)
  attn_wmma_kernel<<<Bb * H * (T / 16), 32, 0, stream>>>(qkvb, attnb, T, H);
  // x2 = attn @ W_proj + b + x
  gemm_bf16_wmma<<<dim3(C / 64, M / 256), 256, 0, stream>>>(
      attnb, bpProj, proj_b, C, C, C, x2, nullptr, x, 2);
  // LN2 -> bf16
  ln_to_bf16_kernel<<<M, 256, 0, stream>>>(x2, ln2_g, ln2_b, abf, C);
  // gate = LN2 @ W_gate + b (fp32)
  gemm_bf16_wmma<<<dim3(HIDP / 64, M / 256), 256, 0, stream>>>(
      abf, bpGate, gate_b, C, HIDP, HID, gateF, nullptr, nullptr, 0);
  // hff = bf16(silu(gate) * (LN2 @ W_up + b))
  gemm_bf16_wmma<<<dim3(HIDP / 64, M / 256), 256, 0, stream>>>(
      abf, bpUp, up_b, C, HIDP, HID, nullptr, hffb, gateF, 3);
  // out = hff @ W_down + b + x2
  gemm_bf16_wmma<<<dim3(C / 64, M / 256), 256, 0, stream>>>(
      hffb, bpDown, down_b, HIDP, C, C, out, nullptr, x2, 2);
}
